// TGNN_embedding_guided_84619445665925
// MI455X (gfx1250) — matrix operand order repaired and checked by reading, not compiled, not run on hardware
//
#include <hip/hip_runtime.h>
#include <hip/hip_bf16.h>

typedef __attribute__((ext_vector_type(16))) _Float16 v16h;
typedef __attribute__((ext_vector_type(8)))  float    v8f;

#define NG_N   320
#define EMB_N  256
#define DIM_N  128
#define T_N    64
#define E2_N   20000
#define EF_N   500000
#define FEAT_N 166

__device__ __forceinline__ float siluf(float x){ return x / (1.0f + __expf(-x)); }
__device__ __forceinline__ float sigmf_(float x){ return 1.0f / (1.0f + __expf(-x)); }

// ---------------------------------------------------------------- elementwise
__global__ void k_fill(float* p, float v, int n){
  int i = blockIdx.x*blockDim.x + threadIdx.x;
  if (i < n) p[i] = v;
}

__global__ void k_deg(const int* __restrict__ ei, float* __restrict__ deg, int E2, int ng){
  int i = blockIdx.x*blockDim.x + threadIdx.x;
  if (i < E2 + ng){
    int gd = (i < E2) ? ei[E2 + i] : (i - E2);
    atomicAdd(&deg[gd], 1.0f);
  }
}

__global__ void k_enorm(const int* __restrict__ ei, const float* __restrict__ deg,
                        float* __restrict__ enorm, int E2, int ng){
  int i = blockIdx.x*blockDim.x + threadIdx.x;
  if (i < E2 + ng){
    int gs = (i < E2) ? ei[i]      : (i - E2);
    int gd = (i < E2) ? ei[E2 + i] : (i - E2);
    enorm[i] = rsqrtf(fmaxf(deg[gs],1.0f)) * rsqrtf(fmaxf(deg[gd],1.0f));
  }
}

// GCN scatter: out[gd] += (xw[gs]) * enorm
__global__ void k_gcn_scatter(const int* __restrict__ ei, const float* __restrict__ xw,
                              const float* __restrict__ enorm, float* __restrict__ out,
                              int E2, int ng, int C){
  int idx = blockIdx.x*blockDim.x + threadIdx.x;
  int total = (E2 + ng) * C;
  if (idx >= total) return;
  int e = idx / C, ch = idx % C;
  int gs = (e < E2) ? ei[e]      : (e - E2);
  int gd = (e < E2) ? ei[E2 + e] : (e - E2);
  atomicAdd(&out[gd*C + ch], xw[gs*C + ch] * enorm[e]);
}

__global__ void k_bias_act(float* __restrict__ x, const float* __restrict__ b, int n, int C, int act){
  int i = blockIdx.x*blockDim.x + threadIdx.x;
  if (i >= n) return;
  float v = x[i] + b[i % C];
  if (act == 1) v = fmaxf(v, 0.0f);
  x[i] = v;
}

__global__ void k_colmax(const float* __restrict__ x, float* __restrict__ out, int R, int C){
  int c = blockIdx.x*blockDim.x + threadIdx.x;
  if (c >= C) return;
  float m = -3.0e38f;
  for (int r = 0; r < R; ++r) m = fmaxf(m, x[r*C + c]);
  out[c] = m;
}

// out[i,j] = a[i]*W[j] + b[j]
__global__ void k_outer(const float* __restrict__ a, const float* __restrict__ W,
                        const float* __restrict__ b, float* __restrict__ out, int n, int dout){
  int idx = blockIdx.x*blockDim.x + threadIdx.x;
  if (idx >= n*dout) return;
  int i = idx / dout, j = idx % dout;
  out[idx] = a[i]*W[j] + b[j];
}

__global__ void k_rowmean(const float* __restrict__ x, float* __restrict__ out, int R, int C){
  int i = blockIdx.x*blockDim.x + threadIdx.x;
  if (i >= R) return;
  float s = 0.0f;
  for (int j = 0; j < C; ++j) s += x[i*C + j];
  out[i] = s / (float)C;
}

__global__ void k_concat2(const float* __restrict__ a, int ca, const float* __restrict__ b, int cb,
                          float* __restrict__ out, int n){
  int C = ca + cb;
  int idx = blockIdx.x*blockDim.x + threadIdx.x;
  if (idx >= n*C) return;
  int i = idx / C, j = idx % C;
  out[idx] = (j < ca) ? a[i*ca + j] : b[i*cb + (j - ca)];
}

__global__ void k_concat3(const float* __restrict__ a, int ca, const float* __restrict__ b, int cb,
                          const float* __restrict__ c, int cc, float* __restrict__ out, int n){
  int C = ca + cb + cc;
  int idx = blockIdx.x*blockDim.x + threadIdx.x;
  if (idx >= n*C) return;
  int i = idx / C, j = idx % C;
  float v;
  if (j < ca)            v = a[i*ca + j];
  else if (j < ca + cb)  v = b[i*cb + (j - ca)];
  else                   v = c[i*cc + (j - ca - cb)];
  out[idx] = v;
}

__global__ void k_pe(const float* __restrict__ t_node, float* __restrict__ pe, int T, int d){
  int i = blockIdx.x, j = threadIdx.x;          // T blocks x d threads
  float tx = t_node[i] * (4000.0f / 1000.0f);
  int half = d >> 1;
  int jj = (j < half) ? j : (j - half);
  float freq = __expf((float)jj * (-logf(10000.0f) / (float)(half - 1)));
  pe[i*d + j] = (j < half) ? sinf(tx*freq) : cosf(tx*freq);
}

__global__ void k_mish(const float* __restrict__ x, float* __restrict__ y, int n){
  int i = blockIdx.x*blockDim.x + threadIdx.x;
  if (i >= n) return;
  float v = x[i];
  float sp = (v > 20.0f) ? v : log1pf(__expf(v));
  y[i] = v * tanhf(sp);
}

__global__ void k_add(const float* __restrict__ a, const float* __restrict__ b,
                      float* __restrict__ c, int n){
  int i = blockIdx.x*blockDim.x + threadIdx.x;
  if (i < n) c[i] = a[i] + b[i];
}

// ---------------------------------------------------------------- attention
__global__ void k_attn_score(const int* __restrict__ ei, const float* __restrict__ q,
                             const float* __restrict__ k, float* __restrict__ score,
                             int E2, int H, int d){
  int idx = blockIdx.x*blockDim.x + threadIdx.x;
  if (idx >= E2*H) return;
  int e = idx / H, h = idx % H;
  int s = ei[e], dd = ei[E2 + e];
  const float* qp = q + ((long long)dd*H + h)*d;   // q[dst]
  const float* kp = k + ((long long)s *H + h)*d;   // k[src]
  float acc = 0.0f;
  for (int j = 0; j < d; ++j) acc += qp[j]*kp[j];
  score[idx] = acc * rsqrtf((float)d);
}

__device__ __forceinline__ void atomicMaxF(float* addr, float val){
  unsigned int* ua = (unsigned int*)addr;
  unsigned int old = *ua, assumed;
  do {
    assumed = old;
    if (__uint_as_float(assumed) >= val) break;
    old = atomicCAS(ua, assumed, __float_as_uint(val));
  } while (old != assumed);
}

__global__ void k_segmax(const int* __restrict__ ei, const float* __restrict__ sc,
                         float* __restrict__ smax, int E2, int H){
  int idx = blockIdx.x*blockDim.x + threadIdx.x;
  if (idx >= E2*H) return;
  int e = idx / H, h = idx % H;
  int dd = ei[E2 + e];
  atomicMaxF(&smax[dd*H + h], sc[idx]);
}

__global__ void k_expden(const int* __restrict__ ei, const float* __restrict__ smax,
                         float* __restrict__ sc, float* __restrict__ den, int E2, int H){
  int idx = blockIdx.x*blockDim.x + threadIdx.x;
  if (idx >= E2*H) return;
  int e = idx / H, h = idx % H;
  int dd = ei[E2 + e];
  float ex = __expf(sc[idx] - smax[dd*H + h]);
  sc[idx] = ex;
  atomicAdd(&den[dd*H + h], ex);
}

__global__ void k_agg(const int* __restrict__ ei, const float* __restrict__ ex,
                      const float* __restrict__ den, const float* __restrict__ v,
                      float* __restrict__ agg, int E2, int H, int d){
  long long idx = (long long)blockIdx.x*blockDim.x + threadIdx.x;
  if (idx >= (long long)E2*H*d) return;
  int j  = (int)(idx % d);
  int eh = (int)(idx / d);
  int h = eh % H, e = eh / H;
  int s = ei[e], dd = ei[E2 + e];
  float alpha = ex[e*H + h] / den[dd*H + h];
  atomicAdd(&agg[((long long)dd*H + h)*d + j], alpha * v[((long long)s*H + h)*d + j]);
}

__global__ void k_tconv_combine(const float* __restrict__ agg, const float* __restrict__ xs,
                                float* __restrict__ out, int n, int H, int d){
  int idx = blockIdx.x*blockDim.x + threadIdx.x;
  if (idx >= n*d) return;
  int i = idx / d, j = idx % d;
  float s = 0.0f;
  for (int h = 0; h < H; ++h) s += agg[(i*H + h)*d + j];
  out[idx] = siluf(s / (float)H + xs[idx]);     // nd = silu(tconv(nd))
}

__global__ void k_pool(const float* __restrict__ x, float* __restrict__ out,
                       int Rin, int Rout, int C){
  int idx = blockIdx.x*blockDim.x + threadIdx.x;
  if (idx >= Rout*C) return;
  int i = idx / C, j = idx % C;
  int s = (i*Rin) / Rout;
  int e = ((i + 1)*Rin + Rout - 1) / Rout;
  float acc = 0.0f;
  for (int r = s; r < e; ++r) acc += x[r*C + j];
  out[idx] = acc / (float)(e - s);
}

__global__ void k_gru(const float* __restrict__ gi, const float* __restrict__ gh,
                      const float* __restrict__ h, float* __restrict__ hout, int n, int d){
  int idx = blockIdx.x*blockDim.x + threadIdx.x;
  if (idx >= n*d) return;
  int i = idx / d, j = idx % d;
  const float* gip = gi + i*3*d;
  const float* ghp = gh + i*3*d;
  float r = sigmf_(gip[j]       + ghp[j]);
  float z = sigmf_(gip[d + j]   + ghp[d + j]);
  float c = tanhf (gip[2*d + j] + r*ghp[2*d + j]);
  hout[idx] = (1.0f - z)*c + z*h[idx];
}

__global__ void k_gemv(const float* __restrict__ x, const float* __restrict__ w,
                       const float* __restrict__ b, float* __restrict__ out, int n, int K){
  int i = blockIdx.x*blockDim.x + threadIdx.x;
  if (i >= n) return;
  float s = 0.0f;
  for (int k = 0; k < K; ++k) s += x[i*K + k]*w[k];
  out[i] = s + b[0];
}

// ---------------------------------------------------------------- WMMA GEMM
// One wave computes one 16x16 tile of C = act(A(MxK) @ W(KxN) + bias).
// M, N must be multiples of 16; K arbitrary (zero-padded). fp32 in, f16 WMMA, fp32 acc.
template<int ACT>
__global__ void k_gemm_wmma(const float* __restrict__ A, const float* __restrict__ W,
                            const float* __restrict__ bias, float* __restrict__ C,
                            int M, int N, int K){
  int lane = threadIdx.x & 31;
  int lid  = lane & 15;
  int half = lane >> 4;
  int tm = blockIdx.x * 16;
  int tn = blockIdx.y * 16;
  int r = tm + lid;      // A row this lane gathers
  int c = tn + lid;      // B/C column this lane owns
  v8f acc = {};
  for (int k0 = 0; k0 < K; k0 += 32){
    v16h a, b;
    #pragma unroll
    for (int v = 0; v < 8; ++v){
      // A k order used here (contiguous, matches B): lanes 0-15 K=0..15, lanes 16-31 K=16..31
      int ka = k0 + half*16 + 2*v;
      a[2*v]   = (ka   < K) ? (_Float16)A[r*K + ka]     : (_Float16)0.0f;
      a[2*v+1] = (ka+1 < K) ? (_Float16)A[r*K + ka + 1] : (_Float16)0.0f;
      // B k order used here (contiguous, matches A): lanes 0-15 K=0..15, lanes 16-31 K=16..31, col = lid
      int kb = k0 + half*16 + 2*v;
      b[2*v]   = (kb   < K) ? (_Float16)W[(long long)kb*N + c]       : (_Float16)0.0f;
      b[2*v+1] = (kb+1 < K) ? (_Float16)W[(long long)(kb + 1)*N + c] : (_Float16)0.0f;
    }
    acc = __builtin_amdgcn_wmma_f32_16x16x32_f16(false, a, false, b, (short)0, acc, false, false);
  }
  float bv = bias ? bias[c] : 0.0f;
  #pragma unroll
  for (int rr = 0; rr < 8; ++rr){
    float x = acc[rr] + bv;                     // C VGPR rr -> row rr (lanes<16) / rr+8
    if (ACT == 1) x = fmaxf(x, 0.0f);
    else if (ACT == 2) x = siluf(x);
    C[(tm + half*8 + rr)*N + c] = x;
  }
}

// ---------------------------------------------------------------- fused edge classifier
// out[e] = silu((nodes[row%n0]+nodes[col%n0]) @ W1 + b1) @ W2 + b2
// W1 staged in LDS as f16 (64 KB); 16 edges x 128 hidden per wave via 8 WMMA frags,
// 128->2 projection done in-register with half-wave shuffle reductions.
__global__ __launch_bounds__(256) void k_edge_class(
    const int* __restrict__ fei, const float* __restrict__ nodes,
    const float* __restrict__ W1, const float* __restrict__ b1,
    const float* __restrict__ W2, const float* __restrict__ b2,
    float* __restrict__ out, int EF, int n0){
  __shared__ _Float16 sW1[256*128];
  int tid = threadIdx.x;
  for (int i = tid; i < 256*128; i += 256) sW1[i] = (_Float16)W1[i];
  __syncthreads();

  int lane = tid & 31, wave = tid >> 5;
  int lid  = lane & 15, half = lane >> 4;
  float b2v0 = b2[0], b2v1 = b2[1];
  // per-lane epilogue weights: column cn = 16*j + lid, j = 0..7
  float bR[8], w0[8], w1[8];
  #pragma unroll
  for (int j = 0; j < 8; ++j){
    int cn = j*16 + lid;
    bR[j] = b1[cn];
    w0[j] = W2[cn*2 + 0];
    w1[j] = W2[cn*2 + 1];
  }

  int ngroups = EF / 16;
  for (int g = blockIdx.x*8 + wave; g < ngroups; g += gridDim.x*8){
    int e  = g*16 + lid;
    int ra = fei[e]      % n0;
    int rb = fei[EF + e] % n0;
    const float* pa = nodes + ra*256;
    const float* pb = nodes + rb*256;
    v8f acc[8] = {};
    #pragma unroll
    for (int k0 = 0; k0 < 256; k0 += 32){
      v16h a;
      #pragma unroll
      for (int v = 0; v < 8; ++v){
        int ka = k0 + half*16 + 2*v;
        a[2*v]   = (_Float16)(pa[ka]     + pb[ka]);
        a[2*v+1] = (_Float16)(pa[ka + 1] + pb[ka + 1]);
      }
      #pragma unroll
      for (int j = 0; j < 8; ++j){
        v16h bf;
        int cn = j*16 + lid;
        #pragma unroll
        for (int v = 0; v < 8; ++v){
          int kb = k0 + half*16 + 2*v;
          bf[2*v]   = sW1[kb*128 + cn];
          bf[2*v+1] = sW1[(kb + 1)*128 + cn];
        }
        acc[j] = __builtin_amdgcn_wmma_f32_16x16x32_f16(false, a, false, bf, (short)0, acc[j], false, false);
      }
    }
    #pragma unroll
    for (int rr = 0; rr < 8; ++rr){
      float p0 = 0.0f, p1 = 0.0f;
      #pragma unroll
      for (int j = 0; j < 8; ++j){
        float hv = siluf(acc[j][rr] + bR[j]);
        p0 += hv * w0[j];
        p1 += hv * w1[j];
      }
      #pragma unroll
      for (int m = 1; m < 16; m <<= 1){        // reduce within each 16-lane half
        p0 += __shfl_xor(p0, m, 32);
        p1 += __shfl_xor(p1, m, 32);
      }
      if (lid == 0){
        int em = g*16 + half*8 + rr;
        out[em*2 + 0] = p0 + b2v0;
        out[em*2 + 1] = p1 + b2v1;
      }
    }
  }
}

// ---------------------------------------------------------------- host
static inline int cdiv(int a, int b){ return (a + b - 1) / b; }

extern "C" void kernel_launch(void* const* d_in, const int* in_sizes, int n_in,
                              void* d_out, int out_size, void* d_ws, size_t ws_size,
                              hipStream_t stream){
  (void)in_sizes; (void)n_in; (void)out_size; (void)ws_size;
  auto F = [&](int i){ return (const float*)d_in[i]; };
  const int*   ei   = (const int*)d_in[0];     // edge_index (2,20000)
  const int*   fei  = (const int*)d_in[1];     // full_edge_index (2,500000)
  const float* feat = F(2);
  const float* lna  = F(3);                    // log_node_attr (320,2)
  const float* tnod = F(4);
  const float* emb  = F(6);
  // params (dict insertion order, layers depth-first)
  const float *gae_W1=F(7), *gae_b1=F(8), *gae_W2=F(9), *gae_b2=F(10);
  const float *emb_t_W=F(11), *emb_t_b=F(12), *emb_0_W=F(13), *emb_0_b=F(14);
  const float *node_in_W=F(15), *node_in_b=F(16);
  const float *mlp_W1=F(17), *mlp_b1=F(18), *mlp_W2=F(19), *mlp_b2=F(20);
  const float *global_W=F(21), *global_b=F(22);
  const float *ctx_W1=F(23), *ctx_b1=F(24), *ctx_W2=F(25), *ctx_b2=F(26);
  const float *node_mlp_W=F(27), *node_mlp_b=F(28);
  const int LBASE = 29;                        // 4 layers x 10 leaves
  const float *gru_Wih=F(69), *gru_Whh=F(70), *gru_bih=F(71), *gru_bhh=F(72);
  const float *ctx_node_W=F(73), *ctx_node_b=F(74);
  const float *node_out_W=F(75), *node_out_b=F(76);
  const float *fin_W1=F(77), *fin_b1=F(78), *fin_W2=F(79), *fin_b2=F(80);

  float* ws = (float*)d_ws;
  size_t off = 0;
  auto alloc = [&](size_t n){ float* p = ws + off; off += (n + 63) & ~(size_t)63; return p; };
  float *deg=alloc(NG_N), *enorm=alloc(E2_N+NG_N);
  float *xw=alloc(NG_N*EMB_N), *h1=alloc(NG_N*EMB_N), *h1w=alloc(NG_N*EMB_N), *nodes_t=alloc(NG_N*EMB_N);
  float *sub_t=alloc(EMB_N), *emb_t=alloc(EMB_N*DIM_N), *emb_0=alloc(EMB_N*DIM_N);
  float *cat=alloc(EMB_N*3*DIM_N), *nodes=alloc(EMB_N*DIM_N);
  float *pe=alloc(T_N*DIM_N), *ttmp=alloc(T_N*EMB_N), *tvec=alloc(T_N*DIM_N);
  float *tmish=alloc(T_N*DIM_N), *temb=alloc(T_N*DIM_N);
  float *rmean=alloc(EMB_N), *contexts=alloc(EMB_N*DIM_N);
  float *hA=alloc(EMB_N*DIM_N), *hB=alloc(EMB_N*DIM_N);
  float *nd=alloc(NG_N*DIM_N), *ndw=alloc(NG_N*EMB_N);
  float *qb=alloc(NG_N*4*DIM_N), *kb=alloc(NG_N*4*DIM_N), *vb=alloc(NG_N*4*DIM_N);
  float *sc=alloc(E2_N*4), *smax=alloc(NG_N*4), *den=alloc(NG_N*4), *agg=alloc(NG_N*4*DIM_N);
  float *tsw=alloc(NG_N*DIM_N), *tcout=alloc(NG_N*DIM_N), *pooled=alloc(EMB_N*DIM_N);
  float *gi=alloc(EMB_N*3*DIM_N), *gh=alloc(EMB_N*3*DIM_N);
  float *nc1=alloc(EMB_N*4*DIM_N), *nc=alloc(EMB_N*DIM_N), *cn=alloc(EMB_N);
  float *nodes2=alloc(EMB_N*EMB_N);

  auto gemm = [&](int act, const float* A, const float* W, const float* b, float* C,
                  int M, int N, int K){
    dim3 g(M/16, N/16);
    if (act == 2)      k_gemm_wmma<2><<<g, 32, 0, stream>>>(A, W, b, C, M, N, K);
    else if (act == 1) k_gemm_wmma<1><<<g, 32, 0, stream>>>(A, W, b, C, M, N, K);
    else               k_gemm_wmma<0><<<g, 32, 0, stream>>>(A, W, b, C, M, N, K);
  };
  auto fill = [&](float* p, float v, int n){ k_fill<<<cdiv(n,256),256,0,stream>>>(p, v, n); };

  // ---- GCN encoder ----
  fill(deg, 0.0f, NG_N);
  k_deg  <<<cdiv(E2_N+NG_N,256),256,0,stream>>>(ei, deg, E2_N, NG_N);
  k_enorm<<<cdiv(E2_N+NG_N,256),256,0,stream>>>(ei, deg, enorm, E2_N, NG_N);
  gemm(0, feat, gae_W1, nullptr, xw, NG_N, EMB_N, FEAT_N);
  fill(h1, 0.0f, NG_N*EMB_N);
  k_gcn_scatter<<<cdiv((E2_N+NG_N)*EMB_N,256),256,0,stream>>>(ei, xw, enorm, h1, E2_N, NG_N, EMB_N);
  k_bias_act<<<cdiv(NG_N*EMB_N,256),256,0,stream>>>(h1, gae_b1, NG_N*EMB_N, EMB_N, 1);
  gemm(0, h1, gae_W2, nullptr, h1w, NG_N, EMB_N, EMB_N);
  fill(nodes_t, 0.0f, NG_N*EMB_N);
  k_gcn_scatter<<<cdiv((E2_N+NG_N)*EMB_N,256),256,0,stream>>>(ei, h1w, enorm, nodes_t, E2_N, NG_N, EMB_N);
  k_bias_act<<<cdiv(NG_N*EMB_N,256),256,0,stream>>>(nodes_t, gae_b2, NG_N*EMB_N, EMB_N, 0);
  k_colmax<<<cdiv(EMB_N,256),256,0,stream>>>(nodes_t, sub_t, NG_N, EMB_N);

  // ---- embeddings / node init ----
  k_outer<<<cdiv(EMB_N*DIM_N,256),256,0,stream>>>(sub_t, emb_t_W, emb_t_b, emb_t, EMB_N, DIM_N);
  k_outer<<<cdiv(EMB_N*DIM_N,256),256,0,stream>>>(emb,   emb_0_W, emb_0_b, emb_0, EMB_N, DIM_N);
  k_concat2<<<cdiv(EMB_N*2*DIM_N,256),256,0,stream>>>(emb_0, DIM_N, emb_t, DIM_N, cat, EMB_N);
  gemm(2, cat, node_in_W, node_in_b, nodes, EMB_N, DIM_N, 2*DIM_N);

  // ---- time embedding ----
  k_pe<<<T_N, DIM_N, 0, stream>>>(tnod, pe, T_N, DIM_N);
  gemm(2, pe,   mlp_W1, mlp_b1, ttmp, T_N, EMB_N, DIM_N);
  gemm(0, ttmp, mlp_W2, mlp_b2, tvec, T_N, DIM_N, EMB_N);

  // ---- contexts / h init ----
  k_rowmean<<<cdiv(EMB_N,256),256,0,stream>>>(nodes, rmean, EMB_N, DIM_N);
  k_outer<<<cdiv(EMB_N*DIM_N,256),256,0,stream>>>(rmean, global_W, global_b, contexts, EMB_N, DIM_N);
  hipMemcpyAsync(hA, nodes, EMB_N*DIM_N*sizeof(float), hipMemcpyDeviceToDevice, stream);

  float* hcur = hA; float* hnxt = hB;
  const int HEADS[4] = {4, 4, 4, 1};
  for (int L = 0; L < 4; ++L){
    int H = HEADS[L];
    const float *time_W=F(LBASE+10*L+0), *time_b=F(LBASE+10*L+1);
    const float *Wq=F(LBASE+10*L+2), *bq=F(LBASE+10*L+3);
    const float *Wk=F(LBASE+10*L+4), *bk_=F(LBASE+10*L+5);
    const float *Wv=F(LBASE+10*L+6), *bv=F(LBASE+10*L+7);
    const float *Ws=F(LBASE+10*L+8), *bs=F(LBASE+10*L+9);

    k_mish<<<cdiv(T_N*DIM_N,256),256,0,stream>>>(tvec, tmish, T_N*DIM_N);
    gemm(0, tmish, time_W, time_b, temb, T_N, DIM_N, DIM_N);
    hipMemcpyAsync(nd, nodes, EMB_N*DIM_N*sizeof(float), hipMemcpyDeviceToDevice, stream);
    hipMemcpyAsync(nd + EMB_N*DIM_N, temb, T_N*DIM_N*sizeof(float), hipMemcpyDeviceToDevice, stream);
    gemm(2, nd, node_mlp_W, node_mlp_b, ndw, NG_N, EMB_N, DIM_N);

    // transformer conv
    gemm(0, ndw, Wq, bq,  qb, NG_N, H*DIM_N, EMB_N);
    gemm(0, ndw, Wk, bk_, kb, NG_N, H*DIM_N, EMB_N);
    gemm(0, ndw, Wv, bv,  vb, NG_N, H*DIM_N, EMB_N);
    k_attn_score<<<cdiv(E2_N*H,256),256,0,stream>>>(ei, qb, kb, sc, E2_N, H, DIM_N);
    fill(smax, -3.0e38f, NG_N*H);
    k_segmax<<<cdiv(E2_N*H,256),256,0,stream>>>(ei, sc, smax, E2_N, H);
    fill(den, 0.0f, NG_N*H);
    k_expden<<<cdiv(E2_N*H,256),256,0,stream>>>(ei, smax, sc, den, E2_N, H);
    fill(agg, 0.0f, NG_N*H*DIM_N);
    k_agg<<<cdiv(E2_N*H*DIM_N,256),256,0,stream>>>(ei, sc, den, vb, agg, E2_N, H, DIM_N);
    gemm(0, ndw, Ws, bs, tsw, NG_N, DIM_N, EMB_N);
    k_tconv_combine<<<cdiv(NG_N*DIM_N,256),256,0,stream>>>(agg, tsw, tcout, NG_N, H, DIM_N);
    k_pool<<<cdiv(EMB_N*DIM_N,256),256,0,stream>>>(tcout, pooled, NG_N, EMB_N, DIM_N);

    // GRU
    gemm(0, pooled, gru_Wih, gru_bih, gi, EMB_N, 3*DIM_N, DIM_N);
    gemm(0, hcur,   gru_Whh, gru_bhh, gh, EMB_N, 3*DIM_N, DIM_N);
    k_gru<<<cdiv(EMB_N*DIM_N,256),256,0,stream>>>(gi, gh, hcur, hnxt, EMB_N, DIM_N);
    { float* t = hcur; hcur = hnxt; hnxt = t; }

    // context update
    k_concat2<<<cdiv(EMB_N*2*DIM_N,256),256,0,stream>>>(hcur, DIM_N, contexts, DIM_N, cat, EMB_N);
    gemm(2, cat, ctx_W1, ctx_b1, nc1, EMB_N, 4*DIM_N, 2*DIM_N);
    gemm(0, nc1, ctx_W2, ctx_b2, nc,  EMB_N, DIM_N, 4*DIM_N);
    k_gemv<<<cdiv(EMB_N,256),256,0,stream>>>(nc, ctx_node_W, ctx_node_b, cn, EMB_N, DIM_N);
    k_outer<<<cdiv(EMB_N*DIM_N,256),256,0,stream>>>(cn, global_W, global_b, contexts, EMB_N, DIM_N);
    k_add<<<cdiv(EMB_N*DIM_N,256),256,0,stream>>>(hcur, contexts, nodes, EMB_N*DIM_N);
  }

  // ---- node out ----
  k_concat3<<<cdiv(EMB_N*3*DIM_N,256),256,0,stream>>>(nodes, DIM_N, emb_t, DIM_N, emb_0, DIM_N, cat, EMB_N);
  gemm(2, cat, node_out_W, node_out_b, nodes2, EMB_N, EMB_N, 3*DIM_N);

  // ---- outputs: log_node_attr passthrough + edge classifier ----
  hipMemcpyAsync(d_out, lna, NG_N*2*sizeof(float), hipMemcpyDeviceToDevice, stream);
  float* edge_out = (float*)d_out + NG_N*2;
  k_edge_class<<<512, 256, 0, stream>>>(fei, nodes2, fin_W1, fin_b1, fin_W2, fin_b2,
                                        edge_out, EF_N, EMB_N);
}
